// GConvLSTM_20847771254916
// MI455X (gfx1250) — matrix operand order repaired and checked by reading, not compiled, not run on hardware
//
#include <hip/hip_runtime.h>
#include <hip/hip_bf16.h>
#include <math.h>

// ---------- types ----------
typedef __bf16 bf16_t;
typedef bf16_t        v16bf __attribute__((ext_vector_type(16)));
typedef float         v8f   __attribute__((ext_vector_type(8)));
typedef unsigned short v16us __attribute__((ext_vector_type(16)));
typedef float         v4f   __attribute__((ext_vector_type(4)));
typedef unsigned int  v8u   __attribute__((ext_vector_type(8)));

#define CH 64

__device__ __forceinline__ float sigm(float x) { return 1.0f / (1.0f + __expf(-x)); }

// ---------- small utility kernels ----------
__global__ void fill0_kernel(float* __restrict__ p, long n) {
  long i = (long)blockIdx.x * blockDim.x + threadIdx.x;
  if (i < n) p[i] = 0.0f;
}

__global__ void deg_kernel(const int* __restrict__ row, const int* __restrict__ col,
                           const float* __restrict__ ew, float* __restrict__ deg, int E) {
  int e = blockIdx.x * blockDim.x + threadIdx.x;
  if (e < E) {
    int r = row[e];
    if (r != col[e]) atomicAdd(&deg[r], ew[e]);
  }
}

__global__ void dinv_kernel(float* __restrict__ deg, int N) {
  int i = blockIdx.x * blockDim.x + threadIdx.x;
  if (i < N) {
    float d = deg[i];
    deg[i] = (d > 0.0f) ? rsqrtf(d) : 0.0f;
  }
}

__global__ void wn_kernel(const int* __restrict__ row, const int* __restrict__ col,
                          const float* __restrict__ ew, const float* __restrict__ dinv,
                          float* __restrict__ wn, int E) {
  int e = blockIdx.x * blockDim.x + threadIdx.x;
  if (e < E) {
    int r = row[e], c = col[e];
    wn[e] = (r != c) ? (-ew[e] * dinv[r] * dinv[c]) : 0.0f;
  }
}

// y[row] += wn * x[col]  (64 channels; 16 edges/block, 4 ch/thread)
__global__ void prop_kernel(const int* __restrict__ row, const int* __restrict__ col,
                            const float* __restrict__ wn, const float* __restrict__ x,
                            float* __restrict__ y, int E) {
  int t = threadIdx.x;
  int e = blockIdx.x * 16 + (t >> 4);
  if (e >= E) return;
  float w = wn[e];
  if (w == 0.0f) return;
  int ch = (t & 15) * 4;
  int c = col[e], r = row[e];
  const v4f xv = *reinterpret_cast<const v4f*>(x + (long)c * CH + ch);
  float* yp = y + (long)r * CH + ch;
  atomicAdd(yp + 0, w * xv[0]);
  atomicAdd(yp + 1, w * xv[1]);
  atomicAdd(yp + 2, w * xv[2]);
  atomicAdd(yp + 3, w * xv[3]);
}

// Build bf16 transposed weights: Wt[g][slab][n][k] = W_g[slab][k][n]
__global__ void convert_weights_kernel(const float* __restrict__ w0, const float* __restrict__ w1,
                                       const float* __restrict__ w2, const float* __restrict__ w3,
                                       const float* __restrict__ w4, const float* __restrict__ w5,
                                       const float* __restrict__ w6, const float* __restrict__ w7,
                                       unsigned short* __restrict__ wt) {
  int i = blockIdx.x * blockDim.x + threadIdx.x;      // 8*3*64*64 = 98304
  if (i >= 8 * 3 * 64 * 64) return;
  const float* ptrs[8] = { w0, w1, w2, w3, w4, w5, w6, w7 };
  int g    = i / 12288;
  int rem  = i % 12288;
  int slab = rem / 4096;
  int nk   = rem % 4096;
  int n    = nk / 64;
  int k    = nk % 64;
  bf16_t b = (bf16_t)ptrs[g][slab * 4096 + k * 64 + n];
  wt[i] = __builtin_bit_cast(unsigned short, b);
}

__global__ void pack_bias_kernel(const float* __restrict__ bx0, const float* __restrict__ bx1,
                                 const float* __restrict__ bx2, const float* __restrict__ bx3,
                                 const float* __restrict__ bh0, const float* __restrict__ bh1,
                                 const float* __restrict__ bh2, const float* __restrict__ bh3,
                                 float* __restrict__ bias4) {
  int i = threadIdx.x;                                // 256
  const float* bx[4] = { bx0, bx1, bx2, bx3 };
  const float* bh[4] = { bh0, bh1, bh2, bh3 };
  int g = i >> 6, ch = i & 63;
  bias4[i] = bx[g][ch] + bh[g][ch];
}

// ---------- fused Chebyshev-GEMM + LSTM gates (WMMA bf16) ----------
// Block = 512 threads = 16 wave32s; one block per 16-node strip.
// Wave (g, tcol): gate g in {i,f,c,o}, output col tile tcol (16 cols).
// acc(16x16) = sum over {X,H} streams, 3 Chebyshev slabs, K=32 steps (12 WMMAs).
__global__ __launch_bounds__(512, 1)
void cheb_lstm_kernel(const float* __restrict__ X, const float* __restrict__ H,
                      const float* __restrict__ C,
                      const float* __restrict__ P1x, const float* __restrict__ P2x,
                      const float* __restrict__ P1h, const float* __restrict__ P2h,
                      const unsigned short* __restrict__ Wt,
                      const float* __restrict__ bias4,
                      const float* __restrict__ wci, const float* __restrict__ wcf,
                      const float* __restrict__ wco,
                      const float* __restrict__ bi, const float* __restrict__ bfg,
                      const float* __restrict__ bcg, const float* __restrict__ bo,
                      float* __restrict__ out, int N) {
  __shared__ float pre[4][16][CH];

  const int node0 = blockIdx.x * 16;
  const int lane  = threadIdx.x & 31;
  const int wv    = threadIdx.x >> 5;   // 0..15
  const int g     = wv >> 2;            // gate 0..3
  const int tcol  = wv & 3;             // col tile 0..3
  const int khalf = lane >> 4;          // 0/1
  const int m     = lane & 15;          // A-row / B-col within tile

  int rown = node0 + m;
  if (rown >= N) rown = N - 1;
  const long rowoff = (long)rown * CH;

  v8f acc = {};
  #pragma unroll
  for (int stream = 0; stream < 2; ++stream) {
    const float* S  = stream ? H : X;
    const float* P1 = stream ? P1h : P1x;
    const float* P2 = stream ? P2h : P2x;
    const unsigned short* wtg = Wt + (size_t)((stream * 4 + g) * 3) * 4096;
    #pragma unroll
    for (int slab = 0; slab < 3; ++slab) {
      // feature(k) = ca*pA[k] + cb*pB[k]:  T0=X, T1=L^X, T2=2*L^2X - X
      const float* pA = (slab == 0) ? S : ((slab == 1) ? P1 : P2);
      const float* pB = S;
      const float ca  = (slab == 2) ? 2.0f : 1.0f;
      const float cb  = (slab == 2) ? -1.0f : 0.0f;
      const unsigned short* wts = wtg + slab * 4096;
      #pragma unroll
      for (int st = 0; st < 2; ++st) {
        const int kk = st * 32;
        const int k0 = kk + 16 * khalf;  // A layout: contiguous run [k0,k0+16)
        v4f a0 = *reinterpret_cast<const v4f*>(pA + rowoff + k0);
        v4f a1 = *reinterpret_cast<const v4f*>(pA + rowoff + k0 + 4);
        v4f a2 = *reinterpret_cast<const v4f*>(pA + rowoff + k0 + 8);
        v4f a3 = *reinterpret_cast<const v4f*>(pA + rowoff + k0 + 12);
        v16bf afrag;
        if (cb == 0.0f) {
          #pragma unroll
          for (int i = 0; i < 4; ++i) {
            afrag[i]      = (bf16_t)a0[i];
            afrag[4 + i]  = (bf16_t)a1[i];
            afrag[8 + i]  = (bf16_t)a2[i];
            afrag[12 + i] = (bf16_t)a3[i];
          }
        } else {
          v4f c0 = *reinterpret_cast<const v4f*>(pB + rowoff + k0);
          v4f c1 = *reinterpret_cast<const v4f*>(pB + rowoff + k0 + 4);
          v4f c2 = *reinterpret_cast<const v4f*>(pB + rowoff + k0 + 8);
          v4f c3 = *reinterpret_cast<const v4f*>(pB + rowoff + k0 + 12);
          #pragma unroll
          for (int i = 0; i < 4; ++i) {
            afrag[i]      = (bf16_t)(ca * a0[i] + cb * c0[i]);
            afrag[4 + i]  = (bf16_t)(ca * a1[i] + cb * c1[i]);
            afrag[8 + i]  = (bf16_t)(ca * a2[i] + cb * c2[i]);
            afrag[12 + i] = (bf16_t)(ca * a3[i] + cb * c3[i]);
          }
        }
        // B fragment: lane holds Wt[n = tcol*16+m][16 consecutive k], 32B aligned
        union { v8u u8; v16bf b; } bfrag;
        bfrag.u8 = *reinterpret_cast<const v8u*>(wts + (tcol * 16 + m) * 64 + kk + 16 * khalf);
        acc = __builtin_amdgcn_wmma_f32_16x16x32_bf16(
            false, afrag, false, bfrag.b, (short)0, acc, false, false);
      }
    }
  }

  const float bias = bias4[g * CH + tcol * 16 + m];
  #pragma unroll
  for (int r = 0; r < 8; ++r)
    pre[g][r + 8 * khalf][tcol * 16 + m] = acc[r] + bias;   // D layout: M=r+8*khalf, N=m

  __syncthreads();

  // LSTM elementwise over this 16-node strip (1024 elements, 512 threads)
  for (int idx = threadIdx.x; idx < 16 * CH; idx += blockDim.x) {
    int nn = idx >> 6;
    int ch = idx & 63;
    int gn = node0 + nn;
    if (gn >= N) continue;
    long gidx = (long)gn * CH + ch;
    float c  = C[gidx];
    float pi = pre[0][nn][ch];
    float pf = pre[1][nn][ch];
    float pc = pre[2][nn][ch];
    float po = pre[3][nn][ch];
    float I  = sigm(pi + wci[ch] * c + bi[ch]);
    float F  = sigm(pf + wcf[ch] * c + bfg[ch]);
    float T  = tanhf(pc + bcg[ch]);
    float Cn = F * c + I * T;
    float O  = sigm(po + wco[ch] * Cn + bo[ch]);
    float Hn = O * tanhf(Cn);
    out[gidx] = Hn;                      // H_new
    out[(long)N * CH + gidx] = Cn;       // C_new
  }
}

// ---------- host launcher ----------
extern "C" void kernel_launch(void* const* d_in, const int* in_sizes, int n_in,
                              void* d_out, int out_size, void* d_ws, size_t ws_size,
                              hipStream_t stream) {
  (void)n_in; (void)out_size; (void)ws_size;
  const float* X  = (const float*)d_in[0];
  const int*   ei = (const int*)d_in[1];
  const float* ew = (const float*)d_in[2];
  const float* H  = (const float*)d_in[3];
  const float* C  = (const float*)d_in[4];
  const float* Wx[4]  = { (const float*)d_in[5],  (const float*)d_in[7],
                          (const float*)d_in[9],  (const float*)d_in[11] };
  const float* bcx[4] = { (const float*)d_in[6],  (const float*)d_in[8],
                          (const float*)d_in[10], (const float*)d_in[12] };
  const float* Wh[4]  = { (const float*)d_in[13], (const float*)d_in[15],
                          (const float*)d_in[17], (const float*)d_in[19] };
  const float* bch[4] = { (const float*)d_in[14], (const float*)d_in[16],
                          (const float*)d_in[18], (const float*)d_in[20] };
  const float* wci = (const float*)d_in[21];
  const float* wcf = (const float*)d_in[22];
  const float* wco = (const float*)d_in[23];
  const float* bi  = (const float*)d_in[24];
  const float* bfg = (const float*)d_in[25];
  const float* bcg = (const float*)d_in[26];
  const float* bo  = (const float*)d_in[27];

  const int N = in_sizes[0] / CH;       // 50000
  const int E = in_sizes[2];            // 1600000
  const int* row = ei;
  const int* col = ei + E;

  // workspace layout (f32 elements)
  float* ws = (float*)d_ws;
  size_t o = 0;
  float* deg  = ws + o; o += (size_t)N;
  float* wn   = ws + o; o += (size_t)E;
  float* P1x  = ws + o; o += (size_t)N * CH;
  float* P2x  = ws + o; o += (size_t)N * CH;
  float* P1h  = ws + o; o += (size_t)N * CH;
  float* P2h  = ws + o; o += (size_t)N * CH;
  float* bias4 = ws + o; o += 256;
  unsigned short* Wt = (unsigned short*)(ws + o);   // 8*3*64*64 bf16

  // zero deg + wn + propagation buffers (wn overwritten anyway; contiguous fill)
  long zn = (long)N + (long)E + 4L * N * CH;
  fill0_kernel<<<(int)((zn + 511) / 512), 512, 0, stream>>>(ws, zn);

  deg_kernel<<<(E + 255) / 256, 256, 0, stream>>>(row, col, ew, deg, E);
  dinv_kernel<<<(N + 255) / 256, 256, 0, stream>>>(deg, N);
  wn_kernel<<<(E + 255) / 256, 256, 0, stream>>>(row, col, ew, deg, wn, E);

  int pgrid = (E + 15) / 16;
  prop_kernel<<<pgrid, 256, 0, stream>>>(row, col, wn, X,   P1x, E);
  prop_kernel<<<pgrid, 256, 0, stream>>>(row, col, wn, P1x, P2x, E);
  prop_kernel<<<pgrid, 256, 0, stream>>>(row, col, wn, H,   P1h, E);
  prop_kernel<<<pgrid, 256, 0, stream>>>(row, col, wn, P1h, P2h, E);

  convert_weights_kernel<<<(8 * 3 * 64 * 64 + 255) / 256, 256, 0, stream>>>(
      Wx[0], Wx[1], Wx[2], Wx[3], Wh[0], Wh[1], Wh[2], Wh[3], Wt);
  pack_bias_kernel<<<1, 256, 0, stream>>>(
      bcx[0], bcx[1], bcx[2], bcx[3], bch[0], bch[1], bch[2], bch[3], bias4);

  cheb_lstm_kernel<<<(N + 15) / 16, 512, 0, stream>>>(
      X, H, C, P1x, P2x, P1h, P2h, Wt, bias4,
      wci, wcf, wco, bi, bfg, bcg, bo, (float*)d_out, N);
}
